// get_model_18313740550564
// MI455X (gfx1250) — matrix operand order repaired and checked by reading, not compiled, not run on hardware
//
#include <hip/hip_runtime.h>
#include <hip/hip_bf16.h>

typedef __attribute__((ext_vector_type(16))) _Float16 v16h;
typedef __attribute__((ext_vector_type(8)))  float    v8f;

#define BATCH   2
#define NPOINT  8192
#define CH      64
#define KNB     20
#define NPTS    (BATCH*NPOINT)          // 16384
#define ROWS    (NPTS*KNB)              // 327680

// ---------------------------------------------------------------------------
// Fragment-order storage: every WMMA operand (A: 16x32 f16, B: 32x16 f16) is
// stored as 512 contiguous halfs: P[frag*512 + lane*16 + e], so each lane
// loads its 16 elements with one contiguous 32-byte load (2x b128).
//
// A-operand element map (ISA 16-bit A layout): lane = row + 16*hi,
//   K = 8*hi + (e<8 ? e : e+8),  hi selects {0..7,16..23} vs {8..15,24..31}.
// B-operand element map: col = lane&15, K = 32*ks + 8*(lane>=16) + (e<8 ? e : e+8).
// ---------------------------------------------------------------------------

// position of element (row r, k within 0..31) inside a packed A fragment
__device__ __host__ inline int a_pack_pos(int r, int k) {
  int hi  = (k >> 3) & 1;            // which lane-half holds this K
  int rem = k - (hi << 3);           // in {0..7} u {16..23}
  int e   = (rem < 8) ? rem : (rem - 8);
  return ((r + (hi << 4)) << 4) + e; // lane*16 + e
}

__device__ inline v16h load_frag(const _Float16* __restrict__ P, int frag, int lane) {
  return *(const v16h*)(P + (((size_t)frag) << 9) + (lane << 4));
}

// ---------------- pack f32 weight (K x N, row-major) into B fragments -------
__global__ void pack_b_kernel(const float* __restrict__ W, _Float16* __restrict__ out,
                              int K, int N) {
  int t = blockIdx.x * blockDim.x + threadIdx.x;
  if (t >= K * N) return;
  int kSteps = K >> 5;
  int frag = t >> 9;
  int r    = t & 511;
  int lane = r >> 4;
  int e    = r & 15;
  int ct   = frag / kSteps;
  int ks   = frag - ct * kSteps;
  int col  = ct * 16 + (lane & 15);
  int k    = ks * 32 + ((lane & 16) ? 8 : 0) + (e < 8 ? e : e + 8);
  out[t] = (_Float16)W[k * N + col];
}

// ---------------- input embedding: relu(points @ conv_w + conv_b) ------------
// writes f32 (for head) and f16 in A-fragment order (for qkv WMMA)
__global__ void embed_kernel(const float* __restrict__ pts,
                             const float* __restrict__ cw,
                             const float* __restrict__ cb,
                             float* __restrict__ x, _Float16* __restrict__ xhp) {
  int t = blockIdx.x * blockDim.x + threadIdx.x;   // NPTS*64 threads
  int c = t & 63, p = t >> 6;
  float v = pts[p*3+0]*cw[c] + pts[p*3+1]*cw[64+c] + pts[p*3+2]*cw[128+c] + cb[c];
  v = fmaxf(v, 0.f);
  x[t] = v;
  int tile = p >> 4, r = p & 15, ks = c >> 5, kr = c & 31;
  xhp[(((size_t)tile * 2 + ks) << 9) + a_pack_pos(r, kr)] = (_Float16)v;
}

// ---------------- KNN: streaming top-20 per query (shared across both layers)
__global__ void knn_kernel(const float* __restrict__ pos, int* __restrict__ idx) {
  __shared__ float cp[128 * 3];
  int tid = threadIdx.x;                 // 128 threads
  int qid = blockIdx.x * 128 + tid;      // whole block in same batch (128|8192)
  int b   = qid >> 13;
  float qx = pos[qid*3+0], qy = pos[qid*3+1], qz = pos[qid*3+2];
  float bd[KNB]; int bi[KNB];
#pragma unroll
  for (int k = 0; k < KNB; ++k) { bd[k] = 3.4e38f; bi[k] = 0; }
  for (int t = 0; t < NPOINT/128; ++t) {
    __syncthreads();
    int j = t*128 + tid;
    const float* pp = pos + ((size_t)b*NPOINT + j)*3;
    cp[tid*3+0] = pp[0]; cp[tid*3+1] = pp[1]; cp[tid*3+2] = pp[2];
    __syncthreads();
    for (int u = 0; u < 128; ++u) {
      float dx = qx - cp[u*3+0], dy = qy - cp[u*3+1], dz = qz - cp[u*3+2];
      float d = dx*dx + dy*dy + dz*dz;
      if (d < bd[KNB-1]) {               // strict: matches stable top_k ties
        float cd = d; int ci = t*128 + u;
#pragma unroll
        for (int k = 0; k < KNB; ++k) {
          if (cd < bd[k]) {
            float td = bd[k]; int ti = bi[k];
            bd[k] = cd; bi[k] = ci;
            cd = td; ci = ti;
          }
        }
      }
    }
  }
#pragma unroll
  for (int k = 0; k < KNB; ++k) idx[qid*KNB + k] = bi[k];
}

// ---------------- qkv = x @ qkv_w (16384x64 @ 64x192), WMMA -----------------
__global__ void qkv_gemm_kernel(const _Float16* __restrict__ Xp,
                                const _Float16* __restrict__ Wp,
                                float* __restrict__ OUT) {
  int lane = threadIdx.x & 31;           // block = 32 (one wave)
  int row0 = blockIdx.x * 16;
  int col0 = blockIdx.y * 16;
  v8f acc = {};
#pragma unroll
  for (int ks = 0; ks < 2; ++ks) {
    v16h a = load_frag(Xp, blockIdx.x * 2 + ks, lane);
    v16h b = load_frag(Wp, blockIdx.y * 2 + ks, lane);
    acc = __builtin_amdgcn_wmma_f32_16x16x32_f16(false, a, false, b, (short)0, acc, false, false);
  }
  int col = col0 + (lane & 15);
  int rb  = (lane & 16) ? 8 : 0;
#pragma unroll
  for (int r = 0; r < 8; ++r)
    OUT[(row0 + rb + r)*192 + col] = acc[r];
}

// ---------------- fused rel-pos MLP + inp assembly ---------------------------
__global__ void posemb_kernel(const float* __restrict__ pos,
                              const int* __restrict__ nbidx,
                              const float* __restrict__ qkv,
                              const float* __restrict__ pw1, const float* __restrict__ pb1,
                              const _Float16* __restrict__ pw2p, const float* __restrict__ pb2,
                              _Float16* __restrict__ inp) {
  __shared__ float relS[16][3];
  __shared__ int rB[16], rN[16], rJ[16];
  __shared__ __align__(32) _Float16 h1p[2 * 512];   // 2 A-fragments (K=64)
  int lane = threadIdx.x;                // block = 32 (one wave)
  int tile = blockIdx.x;
  if (lane < 16) {
    int g = tile*16 + lane;
    int b = g / (NPOINT*KNB);
    int rem = g - b*(NPOINT*KNB);
    int n = rem / KNB;
    int kk = rem - n*KNB;
    int j = nbidx[(b*NPOINT + n)*KNB + kk];
    rB[lane] = b; rN[lane] = n; rJ[lane] = j;
    const float* pq = pos + ((size_t)b*NPOINT + n)*3;
    const float* pj = pos + ((size_t)b*NPOINT + j)*3;
    relS[lane][0] = pq[0]-pj[0];
    relS[lane][1] = pq[1]-pj[1];
    relS[lane][2] = pq[2]-pj[2];
  }
  __syncthreads();
  for (int t = lane; t < 16*64; t += 32) {
    int r = t >> 6, c = t & 63;
    float v = relS[r][0]*pw1[c] + relS[r][1]*pw1[64+c] + relS[r][2]*pw1[128+c] + pb1[c];
    h1p[((c >> 5) << 9) + a_pack_pos(r, c & 31)] = (_Float16)fmaxf(v, 0.f);
    int b = rB[r], n = rN[r], j = rJ[r];
    float q  = qkv[((size_t)b*NPOINT + n)*192 + c];
    float kv = qkv[((size_t)b*NPOINT + j)*192 + 64 + c];
    int g = tile*16 + r;
    inp[(size_t)g*128 + c] = (_Float16)(q - kv);
  }
  __syncthreads();
#pragma unroll
  for (int ct = 0; ct < 4; ++ct) {
    v8f acc = {};
#pragma unroll
    for (int ks = 0; ks < 2; ++ks) {
      v16h a = *(const v16h*)(&h1p[(ks << 9) + (lane << 4)]);
      v16h b = load_frag(pw2p, ct * 2 + ks, lane);
      acc = __builtin_amdgcn_wmma_f32_16x16x32_f16(false, a, false, b, (short)0, acc, false, false);
    }
    int col = ct*16 + (lane & 15);
    int rb  = (lane & 16) ? 8 : 0;
#pragma unroll
    for (int r = 0; r < 8; ++r) {
      int g = tile*16 + rb + r;
      inp[(size_t)g*128 + 64 + col] = (_Float16)(acc[r] + pb2[col]);
    }
  }
}

// ---------------- fused attn MLP: relu(inp@aw1+ab1)@aw2+ab2 -> sim ----------
// 64 rows/block (4 row-tiles staged in LDS in fragment order), 256 hidden
// cols across 8 waves (2 col-tiles each). Each wave loads its 4 B-fragments
// ONCE into registers and reuses them across the 4 row-tiles (4x less weight
// traffic). 256->1 dot fused: relu, scale by aw2, 16-lane shfl reduce,
// per-row accumulate in LDS.
__global__ void attn_kernel(const _Float16* __restrict__ inp,
                            const _Float16* __restrict__ aw1p,
                            const float* __restrict__ ab1,
                            const float* __restrict__ aw2,
                            const float* __restrict__ ab2,
                            float* __restrict__ sim) {
  __shared__ __align__(32) _Float16 Ap[4 * 4 * 512]; // 4 row-tiles x 4 k-frags
  __shared__ float simred[64];
  int tid  = threadIdx.x;                // block = 256 (8 waves)
  int tile = blockIdx.x;                 // 64-row super-tile
  const _Float16* src = inp + (size_t)tile * 64 * 128;
  // warm L2 for the next super-tile's A data (global_prefetch path)
  if (tid < 64) __builtin_prefetch(src + 64*128 + tid*128, 0, 1);
  for (int i = tid; i < 64*128; i += 256) {
    int row = i >> 7, k = i & 127;
    int rt = row >> 4, r = row & 15;
    Ap[rt*2048 + ((k >> 5) << 9) + a_pack_pos(r, k & 31)] = src[i];
  }
  if (tid < 64) simred[tid] = 0.f;
  __syncthreads();
  int wave = tid >> 5, lane = tid & 31;
  float contrib[4][8];
#pragma unroll
  for (int rt = 0; rt < 4; ++rt)
#pragma unroll
    for (int r = 0; r < 8; ++r) contrib[rt][r] = 0.f;
#pragma unroll
  for (int cti = 0; cti < 2; ++cti) {
    int ct  = wave*2 + cti;
    int col = ct*16 + (lane & 15);
    v16h bf[4];
#pragma unroll
    for (int ks = 0; ks < 4; ++ks)
      bf[ks] = load_frag(aw1p, ct * 4 + ks, lane);
    float b1 = ab1[col], w2 = aw2[col];
#pragma unroll
    for (int rt = 0; rt < 4; ++rt) {
      v8f acc = {};
#pragma unroll
      for (int ks = 0; ks < 4; ++ks) {
        v16h a = *(const v16h*)(&Ap[rt*2048 + (ks << 9) + (lane << 4)]);
        acc = __builtin_amdgcn_wmma_f32_16x16x32_f16(false, a, false, bf[ks], (short)0, acc, false, false);
      }
#pragma unroll
      for (int r = 0; r < 8; ++r)
        contrib[rt][r] += fmaxf(acc[r] + b1, 0.f) * w2;
    }
  }
#pragma unroll
  for (int m = 1; m < 16; m <<= 1)
#pragma unroll
    for (int rt = 0; rt < 4; ++rt)
#pragma unroll
      for (int r = 0; r < 8; ++r)
        contrib[rt][r] += __shfl_xor(contrib[rt][r], m, 32);
  if ((lane & 15) == 0) {
    int rb = (lane & 16) ? 8 : 0;       // lane0 -> rows 0..7, lane16 -> rows 8..15
#pragma unroll
    for (int rt = 0; rt < 4; ++rt)
#pragma unroll
      for (int r = 0; r < 8; ++r)
        atomicAdd(&simred[rt*16 + rb + r], contrib[rt][r]);
  }
  __syncthreads();
  if (tid < 64) sim[tile*64 + tid] = simred[tid] + ab2[0];
}

// ---------------- softmax over K + weighted sum of (v[nb] + emb) ------------
__global__ void aggregate_kernel(const float* __restrict__ sim,
                                 const float* __restrict__ qkv,
                                 const _Float16* __restrict__ inp,
                                 const int* __restrict__ nbidx,
                                 float* __restrict__ xout,
                                 _Float16* __restrict__ xhp) {
  __shared__ float s[KNB];
  __shared__ int jj[KNB];
  int bn = blockIdx.x;                   // point index, block = 64 (channels)
  int c  = threadIdx.x;
  if (c < KNB) {
    s[c]  = sim[bn*KNB + c];
    jj[c] = nbidx[bn*KNB + c];
  }
  __syncthreads();
  float m = -3.4e38f;
#pragma unroll
  for (int k = 0; k < KNB; ++k) m = fmaxf(m, s[k]);
  float w[KNB]; float tot = 0.f;
#pragma unroll
  for (int k = 0; k < KNB; ++k) { w[k] = __expf(s[k] - m); tot += w[k]; }
  float inv = 1.f / tot;
  int b = bn >> 13;
  float acc = 0.f;
#pragma unroll
  for (int k = 0; k < KNB; ++k) {
    int j = jj[k];
    float vv  = qkv[((size_t)b*NPOINT + j)*192 + 128 + c];
    float emb = (float)inp[((size_t)bn*KNB + k)*128 + 64 + c];
    acc += w[k] * inv * (vv + emb);
  }
  xout[bn*64 + c] = acc;
  int tile = bn >> 4, r = bn & 15;
  xhp[(((size_t)tile * 2 + (c >> 5)) << 9) + a_pack_pos(r, c & 31)] = (_Float16)acc;
}

// ---------------- maxpool over N + fc1(relu) + fc3 --------------------------
__global__ void head_kernel(const float* __restrict__ x,
                            const float* __restrict__ f1w, const float* __restrict__ f1b,
                            const float* __restrict__ f3w, const float* __restrict__ f3b,
                            float* __restrict__ out) {
  __shared__ float red[256];
  __shared__ float mx[64];
  __shared__ float h[32];
  int b = blockIdx.x, tid = threadIdx.x; // block = 256
  int c = tid & 63, part = tid >> 6;
  float m = -3.4e38f;
  for (int n = part; n < NPOINT; n += 4)
    m = fmaxf(m, x[((size_t)b*NPOINT + n)*64 + c]);
  red[tid] = m;
  __syncthreads();
  if (tid < 64)
    mx[c] = fmaxf(fmaxf(red[c], red[c+64]), fmaxf(red[c+128], red[c+192]));
  __syncthreads();
  if (tid < 32) {
    float a = f1b[tid];
    for (int cc = 0; cc < 64; ++cc) a += mx[cc] * f1w[cc*32 + tid];
    h[tid] = fmaxf(a, 0.f);
  }
  __syncthreads();
  if (tid < 3) {
    float a = f3b[tid];
    for (int cc = 0; cc < 32; ++cc) a += h[cc] * f3w[cc*3 + tid];
    out[b*3 + tid] = a;
  }
}

// ---------------- host orchestration ----------------------------------------
extern "C" void kernel_launch(void* const* d_in, const int* in_sizes, int n_in,
                              void* d_out, int out_size, void* d_ws, size_t ws_size,
                              hipStream_t stream) {
  (void)in_sizes; (void)n_in; (void)out_size; (void)ws_size;
  const float* pts    = (const float*)d_in[0];
  const float* conv_w = (const float*)d_in[1];
  const float* conv_b = (const float*)d_in[2];
  const float* fc1w   = (const float*)d_in[21];
  const float* fc1b   = (const float*)d_in[22];
  const float* fc3w   = (const float*)d_in[23];
  const float* fc3b   = (const float*)d_in[24];

  char* ws = (char*)d_ws;
  size_t off = 0;
  auto carve = [&](size_t bytes) -> void* {
    void* p = ws + off;
    off += (bytes + 255) & ~(size_t)255;
    return p;
  };
  float*     x     = (float*)     carve((size_t)NPTS*64*4);
  _Float16*  xhp   = (_Float16*)  carve((size_t)NPTS*64*2);
  float*     qkv   = (float*)     carve((size_t)NPTS*192*4);
  int*       nbidx = (int*)       carve((size_t)NPTS*KNB*4);
  _Float16*  inp   = (_Float16*)  carve((size_t)ROWS*128*2);
  float*     sim   = (float*)     carve((size_t)ROWS*4);
  _Float16*  qkvwp = (_Float16*)  carve((size_t)64*192*2);
  _Float16*  pw2p  = (_Float16*)  carve((size_t)64*64*2);
  _Float16*  aw1p  = (_Float16*)  carve((size_t)128*256*2);

  embed_kernel<<<NPTS*64/256, 256, 0, stream>>>(pts, conv_w, conv_b, x, xhp);
  knn_kernel<<<NPTS/128, 128, 0, stream>>>(pts, nbidx);   // pos identical both layers

  for (int L = 0; L < 2; ++L) {
    const float* qkvw = (const float*)d_in[3 + L*9 + 0];
    const float* pw1  = (const float*)d_in[3 + L*9 + 1];
    const float* pb1  = (const float*)d_in[3 + L*9 + 2];
    const float* pw2  = (const float*)d_in[3 + L*9 + 3];
    const float* pb2  = (const float*)d_in[3 + L*9 + 4];
    const float* aw1  = (const float*)d_in[3 + L*9 + 5];
    const float* ab1  = (const float*)d_in[3 + L*9 + 6];
    const float* aw2  = (const float*)d_in[3 + L*9 + 7];
    const float* ab2  = (const float*)d_in[3 + L*9 + 8];

    pack_b_kernel<<<(64*192 + 255)/256, 256, 0, stream>>>(qkvw, qkvwp, 64, 192);
    pack_b_kernel<<<(64*64  + 255)/256, 256, 0, stream>>>(pw2,  pw2p,  64, 64);
    pack_b_kernel<<<(128*256+ 255)/256, 256, 0, stream>>>(aw1,  aw1p, 128, 256);

    qkv_gemm_kernel<<<dim3(NPTS/16, 12), 32, 0, stream>>>(xhp, qkvwp, qkv);
    posemb_kernel<<<ROWS/16, 32, 0, stream>>>(pts, nbidx, qkv, pw1, pb1, pw2p, pb2, inp);
    attn_kernel<<<ROWS/64, 256, 0, stream>>>(inp, aw1p, ab1, aw2, ab2, sim);
    aggregate_kernel<<<NPTS, 64, 0, stream>>>(sim, qkv, inp, nbidx, x, xhp);
  }

  head_kernel<<<BATCH, 256, 0, stream>>>(x, fc1w, fc1b, fc3w, fc3b, (float*)d_out);
}
